// DecodeYoloV3_22694607192623
// MI455X (gfx1250) — 8-bit matrix operand order repaired and checked by reading, compiled, not run on hardware
//
#include <hip/hip_runtime.h>
#include <hip/hip_bf16.h>


#define N_TOT   10647
#define N_PAD   10656
#define NT      666
#define NW      333
#define SORT_M  16384
#define CONF_THR 0.5f
#define IOU_THR  0.5f

typedef __attribute__((ext_vector_type(16))) int   v16i;
typedef __attribute__((ext_vector_type(8)))  float v8f;

#define FP8_ONE 0x38u

__device__ const float c_aw[9] = {116.f, 156.f, 373.f, 30.f, 62.f, 59.f, 10.f, 16.f, 33.f};
__device__ const float c_ah[9] = { 90.f, 198.f, 326.f, 61.f, 45.f, 119.f, 13.f, 30.f, 23.f};

__device__ __forceinline__ float sigmoidf_(float x) {
    return 1.0f / (1.0f + expf(-x));
}

__global__ void yolo_decode(const float* __restrict__ p0,
                            const float* __restrict__ p1,
                            const float* __restrict__ p2,
                            float* __restrict__ rawBox,
                            float* __restrict__ conf0,
                            int*   __restrict__ cls0) {
    int g = blockIdx.x * blockDim.x + threadIdx.x;
    if (g >= N_TOT) return;

    const float* p; int W, off, head;
    if (g < 507)        { p = p0; W = 13; off = 0;    head = 0; }
    else if (g < 2535)  { p = p1; W = 26; off = 507;  head = 1; }
    else                { p = p2; W = 52; off = 2535; head = 2; }

    int t   = g - off;
    int HW  = W * W;
    int a   = t / HW;
    int rem = t - a * HW;
    int y   = rem / W;
    int x   = rem - y * W;

    const float* base = p + (a * 85) * HW + y * W + x;
    float stride = 416.0f / (float)W;

    float tx = base[0];
    float ty = base[(size_t)HW];
    float tw = base[(size_t)2 * HW];
    float th = base[(size_t)3 * HW];
    float tc = base[(size_t)4 * HW];

    float aw = c_aw[3 * head + a];
    float ah = c_ah[3 * head + a];

    float cx = (sigmoidf_(tx) + (float)x) * stride;
    float cy = (sigmoidf_(ty) + (float)y) * stride;
    float bw = expf(tw) * aw;
    float bh = expf(th) * ah;

    const float* cp = base + (size_t)5 * HW;
    float best = cp[0];
    int   bi   = 0;
#pragma unroll 4
    for (int ch = 1; ch < 80; ++ch) {
        float v = cp[(size_t)ch * HW];
        if (v > best) { best = v; bi = ch; }
    }

    rawBox[4 * g + 0] = cx;
    rawBox[4 * g + 1] = cy;
    rawBox[4 * g + 2] = bw;
    rawBox[4 * g + 3] = bh;
    conf0[g] = sigmoidf_(tc);
    cls0[g]  = bi;
}

__global__ void yolo_sort(const float* __restrict__ conf0,
                          int* __restrict__ order) {
    extern __shared__ unsigned long long keys[];
    const int T = 1024;
    int tid = threadIdx.x;

    for (int s = tid; s < SORT_M; s += T) {
        unsigned k32;
        if (s < N_TOT) {
            float c  = conf0[s];
            float sc = (c >= CONF_THR) ? c : -1.0f;
            unsigned u = __float_as_uint(sc);
            k32 = (u >> 31) ? u : ~(u | 0x80000000u);
        } else {
            k32 = 0xFFFFFFFFu;
        }
        keys[s] = ((unsigned long long)k32 << 32) | (unsigned)s;
    }

    for (unsigned k = 2; k <= SORT_M; k <<= 1) {
        for (unsigned j = k >> 1; j > 0; j >>= 1) {
            __syncthreads();
            for (int i = tid; i < SORT_M; i += T) {
                int ixj = i ^ (int)j;
                if (ixj > i) {
                    bool up = ((i & k) == 0);
                    unsigned long long va = keys[i];
                    unsigned long long vb = keys[ixj];
                    if (up ? (va > vb) : (va < vb)) {
                        keys[i]   = vb;
                        keys[ixj] = va;
                    }
                }
            }
        }
    }
    __syncthreads();
    for (int s = tid; s < N_PAD; s += T)
        order[s] = (int)(keys[s] & 0xFFFFFFFFu);
}

__global__ void yolo_gather(const int*   __restrict__ order,
                            const float* __restrict__ rawBox,
                            const float* __restrict__ conf0,
                            const int*   __restrict__ cls0,
                            float* __restrict__ sBox,
                            float* __restrict__ sCor,
                            float* __restrict__ sArea,
                            float* __restrict__ sConf,
                            int*   __restrict__ sCls,
                            unsigned char* __restrict__ rowAny) {
    int s = blockIdx.x * blockDim.x + threadIdx.x;
    if (s >= N_PAD) return;

    float cx = 0.f, cy = 0.f, bw = 0.f, bh = 0.f, cf = -2.f;
    int   k  = 255;
    if (s < N_TOT) {
        int idx = order[s];
        if (idx >= 0 && idx < N_TOT) {
            cx = rawBox[4 * idx + 0];
            cy = rawBox[4 * idx + 1];
            bw = rawBox[4 * idx + 2];
            bh = rawBox[4 * idx + 3];
            cf = conf0[idx];
            k  = cls0[idx];
        }
    }
    sBox[4 * s + 0] = cx;
    sBox[4 * s + 1] = cy;
    sBox[4 * s + 2] = bw;
    sBox[4 * s + 3] = bh;
    float hw = 0.5f * bw, hh = 0.5f * bh;
    sCor[4 * s + 0] = cx - hw;
    sCor[4 * s + 1] = cy - hh;
    sCor[4 * s + 2] = cx + hw;
    sCor[4 * s + 3] = cy + hh;
    sArea[s] = bw * bh;
    sConf[s] = cf;
    sCls[s]  = k;
    rowAny[s] = 0;
}


__device__ __forceinline__ v16i onehotA(int c, int hi) {
    int val = (int)(FP8_ONE << ((c & 3) << 3));
    int rA  = ((c >> 4) << 1) | ((c >> 2) & 1);
    bool mine = (((c >> 3) & 1) == hi);
    v16i a;
#pragma unroll
    for (int r = 0; r < 16; ++r)
        a[r] = (mine && rA == r) ? val : 0;
    return a;
}

__device__ __forceinline__ v16i onehotB(int c, int hi) {
    int val = (int)(FP8_ONE << ((c & 3) << 3));
    int rB  = ((c >> 4) << 1) | ((c >> 2) & 1);
    bool mine = (((c >> 3) & 1) == hi);
    v16i b;
#pragma unroll
    for (int r = 0; r < 16; ++r)
        b[r] = (mine && rB == r) ? val : 0;
    return b;
}

__global__ void yolo_pairs(const float* __restrict__ sCor,
                           const float* __restrict__ sArea,
                           const int*   __restrict__ sCls,
                           unsigned short* __restrict__ sup,
                           unsigned char*  __restrict__ rowAny) {
    int lane = threadIdx.x;
    int ti   = blockIdx.x;
    int i0   = ti * 16;
    int hi   = (lane >> 4) & 1;
    int nn   = lane & 15;

    __shared__ float rB[16][5];
    if (lane < 16) {
        int ig = i0 + lane;
        rB[lane][0] = sCor[4 * ig + 0];
        rB[lane][1] = sCor[4 * ig + 1];
        rB[lane][2] = sCor[4 * ig + 2];
        rB[lane][3] = sCor[4 * ig + 3];
        rB[lane][4] = sArea[ig];
    }
    int cRow = sCls[i0 + nn];
    __syncthreads();

    v16i av = onehotA(cRow, hi);

    for (int tj = blockIdx.y; tj < NT; tj += gridDim.y) {
        if (tj < ti) {
            if (lane < 16)
                sup[(size_t)(i0 + lane) * NT + tj] = 0;
            continue;
        }

        int j0 = tj * 16;
        int jg = j0 + nn;

        int tjn = tj + gridDim.y;
        if (tjn < NT)
            __builtin_prefetch(sCor + (size_t)4 * (tjn * 16 + nn), 0, 3);

        float jx1 = sCor[4 * jg + 0];
        float jy1 = sCor[4 * jg + 1];
        float jx2 = sCor[4 * jg + 2];
        float jy2 = sCor[4 * jg + 3];
        float ja  = sArea[jg];
        int cCol  = sCls[jg];

        v16i bv = onehotB(cCol, hi);

        v8f acc = {};
        acc = __builtin_amdgcn_wmma_f32_16x16x128_fp8_fp8(av, bv, (short)0,
                                                          acc, false, false);

#pragma unroll
        for (int r = 0; r < 8; ++r) {
            int m  = r + 8 * hi;
            int ig = i0 + m;
            float ix1 = rB[m][0], iy1 = rB[m][1];
            float ix2 = rB[m][2], iy2 = rB[m][3];
            float ia  = rB[m][4];

            float ltx = fmaxf(ix1, jx1), lty = fmaxf(iy1, jy1);
            float rbx = fminf(ix2, jx2), rby = fminf(iy2, jy2);
            float w = fmaxf(rbx - ltx, 0.0f);
            float h = fmaxf(rby - lty, 0.0f);
            float inter = w * h;
            float uni   = ia + ja - inter;
            bool pred = (acc[r] > 0.5f) &&
                        (inter > IOU_THR * fmaxf(uni, 1e-9f)) &&
                        (jg > ig);
            unsigned msk = __builtin_amdgcn_ballot_w32(pred);

            if (lane == 0) {
                sup[(size_t)(i0 + r)     * NT + tj] = (unsigned short)(msk & 0xFFFFu);
                sup[(size_t)(i0 + r + 8) * NT + tj] = (unsigned short)(msk >> 16);
                if (msk & 0xFFFFu) rowAny[i0 + r]     = 1;
                if (msk >> 16)     rowAny[i0 + r + 8] = 1;
            }
        }
    }
}

__global__ void yolo_nms(const unsigned* __restrict__ sup32,
                         const unsigned char* __restrict__ rowAny,
                         const float* __restrict__ sConf,
                         const float* __restrict__ sBox,
                         const int*   __restrict__ sCls,
                         float* __restrict__ out) {
    __shared__ unsigned removed[NW];
    __shared__ unsigned keepM[NW];
    __shared__ unsigned validM[NW];
    __shared__ unsigned anyM[NW];
    const int T = 1024;
    const int PD = 8;
    int tid = threadIdx.x;

    for (int w = tid; w < NW; w += T) {
        removed[w] = 0u;
        keepM[w]   = 0u;
        unsigned vb = 0u, ab = 0u;
#pragma unroll 4
        for (int b = 0; b < 32; ++b) {
            int s = w * 32 + b;
            if (s < N_TOT) {
                if (sConf[s] >= CONF_THR) vb |= (1u << b);
                if (rowAny[s])            ab |= (1u << b);
            }
        }
        validM[w] = vb;
        anyM[w]   = ab;
    }
    __syncthreads();

    for (int i = 0; i < N_TOT; ++i) {
        unsigned wi = (unsigned)i >> 5;
        unsigned bi = 1u << (i & 31);

        if (tid < 21) {
            int ip = i + PD;
            if (ip < N_TOT) {
                unsigned wp = (unsigned)ip >> 5, bp = 1u << (ip & 31);
                if ((validM[wp] & bp) && (anyM[wp] & bp))
                    __builtin_prefetch((const char*)(sup32 + (size_t)ip * NW) +
                                       (size_t)tid * 64, 0, 3);
            }
        }

        bool keep = (validM[wi] & bi) && !(removed[wi] & bi);
        if (keep) {
            if (tid == 0) keepM[wi] |= bi;
            if (anyM[wi] & bi) {
                const unsigned* row = sup32 + (size_t)i * NW;
                for (int w = tid; w < NW; w += T) removed[w] |= row[w];
                __syncthreads();
            }
        }
    }
    __syncthreads();

    for (int s = tid; s < N_TOT; s += T) {
        bool k = (keepM[(unsigned)s >> 5] >> (s & 31)) & 1u;
        float b0 = 0.f, b1 = 0.f, b2 = 0.f, b3 = 0.f, cf = 0.f, cl = 0.f;
        if (k) {
            b0 = sBox[4 * s + 0];
            b1 = sBox[4 * s + 1];
            b2 = sBox[4 * s + 2];
            b3 = sBox[4 * s + 3];
            cf = sConf[s];
            cl = (float)sCls[s];
        }
        out[6 * s + 0] = b0;
        out[6 * s + 1] = b1;
        out[6 * s + 2] = b2;
        out[6 * s + 3] = b3;
        out[6 * s + 4] = cf;
        out[6 * s + 5] = cl;
    }
}

static inline size_t align256(size_t x) { return (x + 255) & ~(size_t)255; }

extern "C" void kernel_launch(void* const* d_in, const int* in_sizes, int n_in,
                              void* d_out, int out_size, void* d_ws, size_t ws_size,
                              hipStream_t stream) {
    const float* p0 = (const float*)d_in[0];
    const float* p1 = (const float*)d_in[1];
    const float* p2 = (const float*)d_in[2];
    float* out = (float*)d_out;

    char* ws = (char*)d_ws;
    size_t o = 0;
    float* rawBox = (float*)(ws + o); o = align256(o + (size_t)4 * N_PAD * sizeof(float));
    float* conf0  = (float*)(ws + o); o = align256(o + (size_t)N_PAD * sizeof(float));
    int*   cls0   = (int*)  (ws + o); o = align256(o + (size_t)N_PAD * sizeof(int));
    int*   order  = (int*)  (ws + o); o = align256(o + (size_t)N_PAD * sizeof(int));
    float* sBox   = (float*)(ws + o); o = align256(o + (size_t)4 * N_PAD * sizeof(float));
    float* sCor   = (float*)(ws + o); o = align256(o + (size_t)4 * N_PAD * sizeof(float));
    float* sArea  = (float*)(ws + o); o = align256(o + (size_t)N_PAD * sizeof(float));
    float* sConf  = (float*)(ws + o); o = align256(o + (size_t)N_PAD * sizeof(float));
    int*   sCls   = (int*)  (ws + o); o = align256(o + (size_t)N_PAD * sizeof(int));
    unsigned char* rowAny = (unsigned char*)(ws + o);
    o = align256(o + (size_t)N_PAD);
    unsigned short* sup = (unsigned short*)(ws + o);
    o += (size_t)N_PAD * NT * sizeof(unsigned short);
    (void)o; (void)ws_size; (void)in_sizes; (void)n_in; (void)out_size;

    yolo_decode<<<(N_TOT + 127) / 128, 128, 0, stream>>>(p0, p1, p2,
                                                         rawBox, conf0, cls0);

    yolo_sort<<<1, 1024, SORT_M * sizeof(unsigned long long), stream>>>(conf0, order);

    yolo_gather<<<(N_PAD + 127) / 128, 128, 0, stream>>>(order, rawBox, conf0, cls0,
                                                         sBox, sCor, sArea, sConf,
                                                         sCls, rowAny);

    yolo_pairs<<<dim3(NT, 32), 32, 0, stream>>>(sCor, sArea, sCls, sup, rowAny);

    yolo_nms<<<1, 1024, 0, stream>>>((const unsigned*)sup, rowAny,
                                     sConf, sBox, sCls, out);
}
